// MultiScaleChannelTransformerBlock_54743653155447
// MI455X (gfx1250) — compile-verified, matrix operand order checked by reading, not run on hardware
//
#include <hip/hip_runtime.h>
#include <math.h>

typedef __bf16 bf16;
typedef __attribute__((ext_vector_type(16))) __bf16 v16bf;
typedef __attribute__((ext_vector_type(4)))  __bf16 v4bf;
typedef __attribute__((ext_vector_type(8)))  float  v8f;
typedef __attribute__((ext_vector_type(4)))  unsigned int u32x4;
typedef __attribute__((ext_vector_type(8)))  int i32x8;
typedef __attribute__((ext_vector_type(4)))  int i32x4;

#define DIM   128
#define COH   32
#define HID   340
#define HH    256
#define WWID  256
#define HWPX  (HH*WWID)   // 65536
#define NBATCH 2

#if __has_builtin(__builtin_amdgcn_tensor_load_to_lds) && __has_builtin(__builtin_amdgcn_s_wait_tensorcnt)
#define HAVE_TDM 1
#else
#define HAVE_TDM 0
#endif

// ---------------------------------------------------------------------------
// TDM: async 2D tile load (rows x cols of bf16) global -> LDS, row-major both
// sides. D# layout per CDNA5 ISA 8.3/8.4. Issued by a single lane (TDM ignores
// EXEC); completion via s_wait_tensorcnt on the issuing wave.
// ---------------------------------------------------------------------------
__device__ __forceinline__ void tdm_load_tile_bf16(const void* gsrc, unsigned lds_byte_addr,
                                                   unsigned tile_w, unsigned tile_h,
                                                   unsigned long long row_stride_elems,
                                                   unsigned tensor_w, unsigned tensor_h) {
#if HAVE_TDM
  unsigned long long ga = (unsigned long long)gsrc;
  u32x4 g0;
  g0[0] = 1u;                                         // count=1 (valid user descriptor)
  g0[1] = lds_byte_addr;                              // lds_addr
  g0[2] = (unsigned)(ga & 0xFFFFFFFFu);               // global_addr[31:0]
  g0[3] = (unsigned)((ga >> 32) & 0x1FFFFFFu) | (2u << 30);  // global_addr[56:32], type=2
  i32x8 g1;
  g1[0] = (int)(1u << 16);                            // wg_mask=0, data_size=1 (2B)
  g1[1] = (int)((tensor_w & 0xFFFFu) << 16);          // tensor_dim0[15:0]
  g1[2] = (int)((tensor_w >> 16) | ((tensor_h & 0xFFFFu) << 16));   // td0[31:16], td1[15:0]
  g1[3] = (int)((tensor_h >> 16) | (tile_w << 16));   // td1[31:16], tile_dim0
  g1[4] = (int)tile_h;                                // tile_dim1 (tile_dim2 = 0)
  g1[5] = (int)(row_stride_elems & 0xFFFFFFFFu);      // tensor_dim0_stride[31:0]
  g1[6] = (int)((row_stride_elems >> 32) & 0xFFFFu);  // td0_stride[47:32]; td1_stride[15:0]=0
  g1[7] = 0;
  i32x4 gz = {0, 0, 0, 0};
#if defined(__clang_major__) && (__clang_major__ >= 23)
  i32x8 gz8 = {0, 0, 0, 0, 0, 0, 0, 0};
  __builtin_amdgcn_tensor_load_to_lds(g0, g1, gz, gz, gz8, 0);
#else
  __builtin_amdgcn_tensor_load_to_lds(g0, g1, gz, gz, 0);
#endif
#else
  (void)gsrc; (void)lds_byte_addr; (void)tile_w; (void)tile_h;
  (void)row_stride_elems; (void)tensor_w; (void)tensor_h;
#endif
}

__device__ __forceinline__ void tdm_wait0() {
#if HAVE_TDM
  __builtin_amdgcn_s_wait_tensorcnt((short)0);
#endif
}

// ---------------------------------------------------------------------------
// small helper kernels
// ---------------------------------------------------------------------------
__global__ void cvt_bf16_k(const float* __restrict__ in, bf16* __restrict__ out, long n) {
  long i = (long)blockIdx.x * 256 + threadIdx.x;
  if (i < n) out[i] = (bf16)in[i];
}

__global__ void zero_k(float* __restrict__ p, long n) {
  long i = (long)blockIdx.x * 256 + threadIdx.x;
  if (i < n) p[i] = 0.f;
}

// channel LayerNorm (stats over C=128) for one pixel, output bf16 [b][128][HW]
__global__ void ln_bf16_k(const float* __restrict__ Xin, const float* __restrict__ w,
                          const float* __restrict__ b, bf16* __restrict__ Out) {
  long n = (long)blockIdx.x * 256 + threadIdx.x;
  if (n >= HWPX) return;
  const float* x = Xin + (long)blockIdx.z * DIM * HWPX;
  bf16* o = Out + (long)blockIdx.z * DIM * HWPX;
  float s = 0.f, s2 = 0.f;
  for (int c = 0; c < DIM; ++c) { float v = x[(long)c * HWPX + n]; s += v; s2 += v * v; }
  float mu = s * (1.f / DIM);
  float var = s2 * (1.f / DIM) - mu * mu;
  float inv = rsqrtf(var + 1e-5f);
  for (int c = 0; c < DIM; ++c) {
    float v = (x[(long)c * HWPX + n] - mu) * inv * w[c] + b[c];
    o[(long)c * HWPX + n] = (bf16)v;
  }
}

// per-row L2 norm of qk buffer [b][64][Nr] -> invn[b][64]
__global__ void rownorm_k(const float* __restrict__ QK, int Nr, float* __restrict__ invn) {
  const float* row = QK + ((long)blockIdx.z * 64 + blockIdx.x) * Nr;
  float s = 0.f;
  for (long i = threadIdx.x; i < Nr; i += 256) { float v = row[i]; s += v * v; }
  __shared__ float red[256];
  red[threadIdx.x] = s; __syncthreads();
  for (int o = 128; o > 0; o >>= 1) {
    if (threadIdx.x < o) red[threadIdx.x] += red[threadIdx.x + o];
    __syncthreads();
  }
  if (threadIdx.x == 0)
    invn[blockIdx.z * 64 + blockIdx.x] = 1.f / fmaxf(sqrtf(red[0]), 1e-12f);
}

// split-K  S[b][c][d] += sum_n q[c,n]*k[d,n]
#define QKT_CHUNK 16384
__global__ __launch_bounds__(256)
void qkT_k(const float* __restrict__ QK, int Nr, float* __restrict__ S) {
  const float* base = QK + (long)blockIdx.z * 64 * Nr;
  const float* q = base + 32L * Nr;
  const float* k = base;
  float* s = S + (long)blockIdx.z * 32 * 32;
  long c0 = (long)blockIdx.x * QKT_CHUNK;
  long c1 = c0 + QKT_CHUNK; if (c1 > Nr) c1 = Nr;
  __shared__ float qs[32][64];
  __shared__ float ks[32][64];
  int c  = threadIdx.x >> 3;
  int d0 = (threadIdx.x & 7) * 4;
  float a0 = 0, a1 = 0, a2 = 0, a3 = 0;
  for (long n0 = c0; n0 < c1; n0 += 64) {
    for (int i = threadIdx.x; i < 32 * 64; i += 256) {
      int rr = i >> 6, nn = i & 63;
      long g = n0 + nn;
      float qv = 0.f, kv = 0.f;
      if (g < Nr) { qv = q[(long)rr * Nr + g]; kv = k[(long)rr * Nr + g]; }
      qs[rr][nn] = qv; ks[rr][nn] = kv;
    }
    __syncthreads();
    for (int nn = 0; nn < 64; ++nn) {
      float qv = qs[c][nn];
      a0 += qv * ks[d0 + 0][nn]; a1 += qv * ks[d0 + 1][nn];
      a2 += qv * ks[d0 + 2][nn]; a3 += qv * ks[d0 + 3][nn];
    }
    __syncthreads();
  }
  atomicAdd(&s[c * 32 + d0 + 0], a0); atomicAdd(&s[c * 32 + d0 + 1], a1);
  atomicAdd(&s[c * 32 + d0 + 2], a2); atomicAdd(&s[c * 32 + d0 + 3], a3);
}

// softmax + fold: Wfold[b] = sum_i wproj[:,32i:32i+32] @ attn_i @ wv_i
struct FoldParams {
  const float* S;
  const float* invn;
  const float* t[4];
  const float* wv[4];
  const float* bv[4];
  const float* wproj;
  bf16*  Wfold;
  float* bfold;
};

__global__ __launch_bounds__(256)
void fold_k(FoldParams p) {
  const int b = blockIdx.x;
  __shared__ float attn[32][32];
  __shared__ float P[128][32];
  const int tid = threadIdx.x;
  const int m   = tid >> 1;
  const int ci0 = (tid & 1) * 64;
  float acc[64];
  #pragma unroll
  for (int j = 0; j < 64; ++j) acc[j] = 0.f;
  float bacc = 0.f;
  for (int i = 0; i < 4; ++i) {
    const float* Sb  = p.S    + ((long)i * NBATCH + b) * 32 * 32;
    const float* inv = p.invn + ((long)i * NBATCH + b) * 64;
    const float  tt  = *p.t[i];
    if (tid < 32) {
      int c = tid;
      float row[32]; float mx = -1e30f;
      for (int d = 0; d < 32; ++d) {
        float v = Sb[c * 32 + d] * inv[32 + c] * inv[d] * tt;
        row[d] = v; mx = fmaxf(mx, v);
      }
      float ssum = 0.f;
      for (int d = 0; d < 32; ++d) { float e = __expf(row[d] - mx); row[d] = e; ssum += e; }
      float rs = 1.f / ssum;
      for (int d = 0; d < 32; ++d) attn[c][d] = row[d] * rs;
    }
    __syncthreads();
    for (int idx = tid; idx < 128 * 32; idx += 256) {
      int mm = idx >> 5, d = idx & 31;
      float sv = 0.f;
      for (int c = 0; c < 32; ++c) sv += p.wproj[mm * 128 + i * 32 + c] * attn[c][d];
      P[mm][d] = sv;
    }
    __syncthreads();
    for (int d = 0; d < 32; ++d) {
      float pv = P[m][d];
      const float* wvrow = p.wv[i] + d * 128 + ci0;
      #pragma unroll 8
      for (int j = 0; j < 64; ++j) acc[j] += pv * wvrow[j];
      if ((tid & 1) == 0) bacc += pv * p.bv[i][d];
    }
    __syncthreads();
  }
  bf16* Wo = p.Wfold + (long)b * 128 * 128 + (long)m * 128 + ci0;
  for (int j = 0; j < 64; ++j) Wo[j] = (bf16)acc[j];
  if ((tid & 1) == 0) p.bfold[b * 128 + m] = bacc;
}

// depthwise 3x3 (pad 1) + exact-GELU gate
__global__ void dwglu_k(const bf16* __restrict__ F, const float* __restrict__ Wdw,
                        bf16* __restrict__ G) {
  int n = blockIdx.x * 256 + threadIdx.x;
  if (n >= HWPX) return;
  int y = n >> 8, x = n & (WWID - 1);
  int c = blockIdx.y;
  const bf16* Fb = F + (long)blockIdx.z * (2 * HID) * HWPX;
  float d1 = 0.f, d2 = 0.f;
  const float* w1 = Wdw + (long)c * 9;
  const float* w2 = Wdw + (long)(c + HID) * 9;
  const bf16* f1 = Fb + (long)c * HWPX;
  const bf16* f2 = Fb + (long)(c + HID) * HWPX;
  #pragma unroll
  for (int ky = 0; ky < 3; ++ky)
    #pragma unroll
    for (int kx = 0; kx < 3; ++kx) {
      int yy = y + ky - 1, xx = x + kx - 1;
      if (yy >= 0 && yy < HH && xx >= 0 && xx < WWID) {
        long o = (long)yy * WWID + xx;
        d1 += w1[ky * 3 + kx] * (float)f1[o];
        d2 += w2[ky * 3 + kx] * (float)f2[o];
      }
    }
  float g = 0.5f * d1 * (1.f + erff(d1 * 0.70710678118654752f));
  G[((long)blockIdx.z * HID + c) * HWPX + n] = (bf16)(g * d2);
}

// ---------------------------------------------------------------------------
// bf16 WMMA GEMM  C[M,N] = A[M,K] * B[K,N] (+epilogue), tiles 64x128x32.
// A tile staged by the Tensor Data Mover (async, wave0 issues + waits),
// B tile staged transposed by all threads (feeds WMMA B-operand layout).
// ---------------------------------------------------------------------------
enum { EPI_F32 = 0, EPI_BF16 = 1, EPI_X2 = 2, EPI_OUT = 3 };

template <int EPI>
__global__ __launch_bounds__(256)
void gemm_wmma_k(const bf16* __restrict__ A, long sA,
                 const bf16* __restrict__ B, long sB,
                 float* __restrict__ Cf, bf16* __restrict__ Cb, long sC,
                 const float* __restrict__ X, long sX,
                 const bf16* __restrict__ Y, long sY,
                 const float* __restrict__ bias, long sBias,
                 int M, int N, int K, int ldb, int ldc) {
  A += (long)blockIdx.z * sA;
  B += (long)blockIdx.z * sB;
  if (Cf)   Cf   += (long)blockIdx.z * sC;
  if (Cb)   Cb   += (long)blockIdx.z * sC;
  if (X)    X    += (long)blockIdx.z * sX;
  if (Y)    Y    += (long)blockIdx.z * sY;
  if (bias) bias += (long)blockIdx.z * sBias;

  __shared__ bf16 As[2][64][32];
  __shared__ bf16 Bs[2][128][32];

  const int tid = threadIdx.x, lane = tid & 31, wave = tid >> 5;
  const int mw = wave >> 2, nw = wave & 3;
  const int m0 = blockIdx.y * 64, n0 = blockIdx.x * 128;
  const int mrows = (M - m0 > 64) ? 64 : (M - m0);

  v8f acc[2][2] = {};

  auto stageA_tdm = [&](int buf, int k0) {     // async TDM: [mrows x <=32] bf16 tile
    if (tid == 0) {
      int kw = K - k0; if (kw > 32) kw = 32;
      unsigned lds = (unsigned)(unsigned long long)(&As[buf][0][0]);
      tdm_load_tile_bf16(&A[(long)m0 * K + k0], lds, 32u, 64u,
                         (unsigned long long)K, (unsigned)kw, (unsigned)mrows);
    }
  };
  auto stageA_cpu = [&](int buf, int kt) {     // fallback path
    for (int i = tid; i < 64 * 32 / 4; i += 256) {
      int m = i >> 3, k4 = (i & 7) * 4;
      int gm = m0 + m, gk = kt + k4;
      v4bf val = {};
      if (gm < M && gk + 4 <= K) val = *(const v4bf*)&A[(long)gm * K + gk];
      else if (gm < M) {
        for (int j = 0; j < 4; ++j) val[j] = (gk + j < K) ? A[(long)gm * K + gk + j] : (bf16)0.f;
      }
      *(v4bf*)&As[buf][m][k4] = val;
    }
  };
  auto stageA = [&](int buf, int kt) {
    if (HAVE_TDM) stageA_tdm(buf, kt); else stageA_cpu(buf, kt);
  };
  auto stageB = [&](int buf, int kt) {
    for (int i = tid; i < 32 * 128 / 4; i += 256) {
      int k = i >> 5, n4 = (i & 31) * 4;
      int gk = kt + k;
      v4bf val = {};
      if (gk < K && n0 + n4 + 3 < N) val = *(const v4bf*)&B[(long)gk * ldb + n0 + n4];
      else if (gk < K) {
        for (int j = 0; j < 4; ++j) if (n0 + n4 + j < N) val[j] = B[(long)gk * ldb + n0 + n4 + j];
      }
      Bs[buf][n4 + 0][k] = val[0]; Bs[buf][n4 + 1][k] = val[1];
      Bs[buf][n4 + 2][k] = val[2]; Bs[buf][n4 + 3][k] = val[3];
    }
  };

  const int nk = (K + 31) / 32;
  stageA(0, 0); stageB(0, 0);
  if (HAVE_TDM && wave == 0) tdm_wait0();
  __syncthreads();

  const int kb = (lane < 16) ? 0 : 8;
  const int row16 = lane & 15;

  for (int kt = 0; kt < nk; ++kt) {
    const int cur = kt & 1;
    if (kt + 2 < nk) {
      __builtin_prefetch(&B[(long)((kt + 2) * 32) * ldb + n0], 0, 0);
    }
    if (kt + 1 < nk) { stageA(cur ^ 1, (kt + 1) * 32); stageB(cur ^ 1, (kt + 1) * 32); }

    v16bf afrag[2], bfrag[2];
    #pragma unroll
    for (int mi = 0; mi < 2; ++mi) {
      const bf16* ap = &As[cur][mw * 32 + mi * 16 + row16][kb];
      union { v16bf v; v4bf q[4]; } u;
      u.q[0] = *(const v4bf*)(ap + 0);  u.q[1] = *(const v4bf*)(ap + 4);
      u.q[2] = *(const v4bf*)(ap + 16); u.q[3] = *(const v4bf*)(ap + 20);
      afrag[mi] = u.v;
    }
    #pragma unroll
    for (int ni = 0; ni < 2; ++ni) {
      const bf16* bp = &Bs[cur][nw * 32 + ni * 16 + row16][kb];
      union { v16bf v; v4bf q[4]; } u;
      u.q[0] = *(const v4bf*)(bp + 0);  u.q[1] = *(const v4bf*)(bp + 4);
      u.q[2] = *(const v4bf*)(bp + 16); u.q[3] = *(const v4bf*)(bp + 20);
      bfrag[ni] = u.v;
    }
    #pragma unroll
    for (int mi = 0; mi < 2; ++mi)
      #pragma unroll
      for (int ni = 0; ni < 2; ++ni)
        acc[mi][ni] = __builtin_amdgcn_wmma_f32_16x16x32_bf16(
            false, afrag[mi], false, bfrag[ni], (short)0, acc[mi][ni], false, false);
    if (HAVE_TDM && wave == 0) tdm_wait0();
    __syncthreads();
  }

  const int rbase = m0 + mw * 32, cbase = n0 + nw * 32;
  const int mhalf = (lane >> 4) * 8, nl = lane & 15;
  #pragma unroll
  for (int mi = 0; mi < 2; ++mi)
    #pragma unroll
    for (int ni = 0; ni < 2; ++ni)
      #pragma unroll
      for (int e = 0; e < 8; ++e) {
        int r = rbase + mi * 16 + mhalf + e;
        int c = cbase + ni * 16 + nl;
        if (r < M && c < N) {
          float v = acc[mi][ni][e];
          if (bias) v += bias[r];
          long idx = (long)r * ldc + c;
          if (EPI == EPI_F32)       Cf[idx] = v;
          else if (EPI == EPI_BF16) Cb[idx] = (bf16)v;
          else if (EPI == EPI_X2)   Cf[idx] = X[idx] + (float)Y[idx] + v;
          else                      Cf[idx] = X[idx] + v;   // EPI_OUT
        }
      }
}

// ---------------------------------------------------------------------------
// stride-r conv as im2col GEMM (M=64, K=128*r*r), r in {2,4,8}; TDM A-staging
// ---------------------------------------------------------------------------
__global__ __launch_bounds__(256)
void conv_qk_wmma_k(const bf16* __restrict__ Wq, const bf16* __restrict__ Ybf,
                    float* __restrict__ Out, int r) {
  const int r2 = r * r;
  const int K  = DIM * r2;
  const int Wr = WWID / r;
  const int Nr = HWPX / r2;
  const bf16* Y = Ybf + (long)blockIdx.z * DIM * HWPX;
  float* C = Out + (long)blockIdx.z * 64 * Nr;

  __shared__ bf16 As[2][64][32];
  __shared__ bf16 Bs[2][128][32];
  const int tid = threadIdx.x, lane = tid & 31, wave = tid >> 5;
  const int mw = wave >> 2, nw = wave & 3;
  const int n0 = blockIdx.x * 128;
  v8f acc[2][2] = {};

  auto stageA = [&](int buf, int k0) {
    if (HAVE_TDM) {
      if (tid == 0) {
        unsigned lds = (unsigned)(unsigned long long)(&As[buf][0][0]);
        tdm_load_tile_bf16(&Wq[(long)0 * K + k0], lds, 32u, 64u,
                           (unsigned long long)K, 32u, 64u);
      }
    } else {
      for (int i = tid; i < 64 * 32 / 4; i += 256) {
        int m = i >> 3, k4 = (i & 7) * 4;
        int gk = k0 + k4;
        v4bf val = {};
        if (gk + 4 <= K) val = *(const v4bf*)&Wq[(long)m * K + gk];
        *(v4bf*)&As[buf][m][k4] = val;
      }
    }
  };
  auto stageB = [&](int buf, int kt) {      // im2col gather
    for (int i = tid; i < 32 * 128; i += 256) {
      int k = i >> 7, n = i & 127;
      int gk = kt + k, gn = n0 + n;
      bf16 v = (bf16)0.f;
      if (gk < K && gn < Nr) {
        int ci = gk / r2, t2 = gk - ci * r2;
        int ky = t2 / r,  kx = t2 - ky * r;
        int oy = gn / Wr, ox = gn - oy * Wr;
        v = Y[(long)ci * HWPX + (long)(oy * r + ky) * WWID + ox * r + kx];
      }
      Bs[buf][n][k] = v;
    }
  };

  const int nk = K / 32;
  stageA(0, 0); stageB(0, 0);
  if (HAVE_TDM && wave == 0) tdm_wait0();
  __syncthreads();
  const int kb = (lane < 16) ? 0 : 8;
  const int row16 = lane & 15;
  for (int kt = 0; kt < nk; ++kt) {
    const int cur = kt & 1;
    if (kt + 1 < nk) { stageA(cur ^ 1, (kt + 1) * 32); stageB(cur ^ 1, (kt + 1) * 32); }
    v16bf afrag[2], bfrag[2];
    #pragma unroll
    for (int mi = 0; mi < 2; ++mi) {
      const bf16* ap = &As[cur][mw * 32 + mi * 16 + row16][kb];
      union { v16bf v; v4bf q[4]; } u;
      u.q[0] = *(const v4bf*)(ap + 0);  u.q[1] = *(const v4bf*)(ap + 4);
      u.q[2] = *(const v4bf*)(ap + 16); u.q[3] = *(const v4bf*)(ap + 20);
      afrag[mi] = u.v;
    }
    #pragma unroll
    for (int ni = 0; ni < 2; ++ni) {
      const bf16* bp = &Bs[cur][nw * 32 + ni * 16 + row16][kb];
      union { v16bf v; v4bf q[4]; } u;
      u.q[0] = *(const v4bf*)(bp + 0);  u.q[1] = *(const v4bf*)(bp + 4);
      u.q[2] = *(const v4bf*)(bp + 16); u.q[3] = *(const v4bf*)(bp + 20);
      bfrag[ni] = u.v;
    }
    #pragma unroll
    for (int mi = 0; mi < 2; ++mi)
      #pragma unroll
      for (int ni = 0; ni < 2; ++ni)
        acc[mi][ni] = __builtin_amdgcn_wmma_f32_16x16x32_bf16(
            false, afrag[mi], false, bfrag[ni], (short)0, acc[mi][ni], false, false);
    if (HAVE_TDM && wave == 0) tdm_wait0();
    __syncthreads();
  }
  const int rbase = mw * 32, cbase = n0 + nw * 32;
  const int mhalf = (lane >> 4) * 8, nl = lane & 15;
  #pragma unroll
  for (int mi = 0; mi < 2; ++mi)
    #pragma unroll
    for (int ni = 0; ni < 2; ++ni)
      #pragma unroll
      for (int e = 0; e < 8; ++e) {
        int r_ = rbase + mi * 16 + mhalf + e;
        int c_ = cbase + ni * 16 + nl;
        if (c_ < Nr) C[(long)r_ * Nr + c_] = acc[mi][ni][e];
      }
}

// ---------------------------------------------------------------------------
extern "C" void kernel_launch(void* const* d_in, const int* in_sizes, int n_in,
                              void* d_out, int out_size, void* d_ws, size_t ws_size,
                              hipStream_t stream) {
  (void)in_sizes; (void)n_in; (void)out_size; (void)ws_size;
  const float* x     = (const float*)d_in[0];
  const float* n1w   = (const float*)d_in[1];
  const float* n1b   = (const float*)d_in[2];
  const float* n2w   = (const float*)d_in[3];
  const float* n2b   = (const float*)d_in[4];
  const float* wqk[4], *wv[4], *bv[4], *tt[4];
  for (int i = 0; i < 4; ++i) {
    wqk[i] = (const float*)d_in[5 + 4 * i + 0];
    wv[i]  = (const float*)d_in[5 + 4 * i + 1];
    bv[i]  = (const float*)d_in[5 + 4 * i + 2];
    tt[i]  = (const float*)d_in[5 + 4 * i + 3];
  }
  const float* wproj = (const float*)d_in[21];
  const float* wpi   = (const float*)d_in[22];
  const float* wdw   = (const float*)d_in[23];
  const float* wpo   = (const float*)d_in[24];
  float* out = (float*)d_out;

  const int scales[4] = {1, 2, 4, 8};
  char* w = (char*)d_ws; size_t off = 0;
  auto alloc = [&](size_t bytes) -> void* {
    void* p = w + off; off = (off + bytes + 255) & ~(size_t)255; return p;
  };
  bf16* ybf   = (bf16*)alloc((size_t)NBATCH * DIM * HWPX * 2);
  bf16* zbf   = (bf16*)alloc((size_t)NBATCH * DIM * HWPX * 2);
  bf16* wqkb[4];
  for (int i = 0; i < 4; ++i) wqkb[i] = (bf16*)alloc((size_t)64 * DIM * scales[i] * scales[i] * 2);
  bf16* wpib  = (bf16*)alloc((size_t)(2 * HID) * DIM * 2);
  bf16* wpob  = (bf16*)alloc((size_t)DIM * HID * 2);
  float* qkbuf[4];
  for (int i = 0; i < 4; ++i) {
    int Nr = HWPX / (scales[i] * scales[i]);
    qkbuf[i] = (float*)alloc((size_t)NBATCH * 64 * Nr * 4);
  }
  float* invn  = (float*)alloc((size_t)4 * NBATCH * 64 * 4);
  float* Sbuf  = (float*)alloc((size_t)4 * NBATCH * 32 * 32 * 4);
  bf16*  Wfold = (bf16*)alloc((size_t)NBATCH * DIM * DIM * 2);
  float* bfold = (float*)alloc((size_t)NBATCH * DIM * 4);
  float* x2    = (float*)alloc((size_t)NBATCH * DIM * HWPX * 4);
  bf16*  fbuf  = (bf16*)alloc((size_t)NBATCH * (2 * HID) * HWPX * 2);
  bf16*  gbuf  = (bf16*)alloc((size_t)NBATCH * HID * HWPX * 2);

  dim3 blk(256);
  for (int i = 0; i < 4; ++i) {
    long n = 64L * DIM * scales[i] * scales[i];
    cvt_bf16_k<<<dim3((n + 255) / 256), blk, 0, stream>>>(wqk[i], wqkb[i], n);
  }
  { long n = (long)(2 * HID) * DIM; cvt_bf16_k<<<dim3((n + 255) / 256), blk, 0, stream>>>(wpi, wpib, n); }
  { long n = (long)DIM * HID;       cvt_bf16_k<<<dim3((n + 255) / 256), blk, 0, stream>>>(wpo, wpob, n); }

  // LN1 -> y (bf16)
  ln_bf16_k<<<dim3(HWPX / 256, 1, NBATCH), blk, 0, stream>>>(x, n1w, n1b, ybf);

  // qk convs: r=1 is a plain GEMM; r>1 via im2col gather
  gemm_wmma_k<EPI_F32><<<dim3(HWPX / 128, 1, NBATCH), blk, 0, stream>>>(
      wqkb[0], 0L, ybf, (long)DIM * HWPX,
      qkbuf[0], nullptr, (long)64 * HWPX,
      nullptr, 0L, nullptr, 0L, nullptr, 0L,
      64, HWPX, DIM, HWPX, HWPX);
  for (int i = 1; i < 4; ++i) {
    int r = scales[i], Nr = HWPX / (r * r);
    conv_qk_wmma_k<<<dim3(Nr / 128, 1, NBATCH), blk, 0, stream>>>(wqkb[i], ybf, qkbuf[i], r);
  }
  // row norms + raw q.kT
  zero_k<<<dim3((4 * NBATCH * 32 * 32 + 255) / 256), blk, 0, stream>>>(Sbuf, 4L * NBATCH * 32 * 32);
  for (int i = 0; i < 4; ++i) {
    int r = scales[i], Nr = HWPX / (r * r);
    rownorm_k<<<dim3(64, 1, NBATCH), blk, 0, stream>>>(qkbuf[i], Nr, invn + (long)i * NBATCH * 64);
    int nch = (Nr + QKT_CHUNK - 1) / QKT_CHUNK;
    qkT_k<<<dim3(nch, 1, NBATCH), blk, 0, stream>>>(qkbuf[i], Nr, Sbuf + (long)i * NBATCH * 32 * 32);
  }
  // softmax + fold attention into Wfold/bfold
  FoldParams fp;
  fp.S = Sbuf; fp.invn = invn; fp.wproj = wproj; fp.Wfold = Wfold; fp.bfold = bfold;
  for (int i = 0; i < 4; ++i) { fp.t[i] = tt[i]; fp.wv[i] = wv[i]; fp.bv[i] = bv[i]; }
  fold_k<<<dim3(NBATCH), blk, 0, stream>>>(fp);

  // x2 = x + Wfold@y + bfold + y
  gemm_wmma_k<EPI_X2><<<dim3(HWPX / 128, DIM / 64, NBATCH), blk, 0, stream>>>(
      Wfold, (long)DIM * DIM, ybf, (long)DIM * HWPX,
      x2, nullptr, (long)DIM * HWPX,
      x, (long)DIM * HWPX, ybf, (long)DIM * HWPX,
      bfold, DIM, DIM, HWPX, DIM, HWPX, HWPX);

  // LN2 -> z (bf16)
  ln_bf16_k<<<dim3(HWPX / 256, 1, NBATCH), blk, 0, stream>>>(x2, n2w, n2b, zbf);

  // f = wpi @ z   (M=680, bf16 out)
  gemm_wmma_k<EPI_BF16><<<dim3(HWPX / 128, (2 * HID + 63) / 64, NBATCH), blk, 0, stream>>>(
      wpib, 0L, zbf, (long)DIM * HWPX,
      nullptr, fbuf, (long)(2 * HID) * HWPX,
      nullptr, 0L, nullptr, 0L, nullptr, 0L,
      2 * HID, HWPX, DIM, HWPX, HWPX);

  // depthwise 3x3 + gelu gate -> g (bf16)
  dwglu_k<<<dim3(HWPX / 256, HID, NBATCH), blk, 0, stream>>>(fbuf, wdw, gbuf);

  // out = x2 + wpo @ g   (K=340)
  gemm_wmma_k<EPI_OUT><<<dim3(HWPX / 128, DIM / 64, NBATCH), blk, 0, stream>>>(
      wpob, 0L, gbuf, (long)HID * HWPX,
      out, nullptr, (long)DIM * HWPX,
      x2, (long)DIM * HWPX, nullptr, 0L, nullptr, 0L,
      DIM, HWPX, HID, HWPX, HWPX);
}
